// MultiHeadSelfAttention_90134183674627
// MI455X (gfx1250) — compile-verified, matrix operand order checked by reading, not run on hardware
//
#include <hip/hip_runtime.h>

#define HIDDEN   1024
#define HEADS    16
#define HEAD_DIM 64
#define BATCH    4
#define SEQ      2048

typedef __attribute__((ext_vector_type(16))) __bf16 v16bf;
typedef __attribute__((ext_vector_type(8)))  __bf16 v8bf;
typedef __attribute__((ext_vector_type(4)))  __bf16 v4bf;
typedef __attribute__((ext_vector_type(8)))  float  v8f;
typedef __attribute__((ext_vector_type(4)))  float  v4f;

static __device__ inline __bf16 f2bf(float f) {
  union { float f; unsigned int u; } v; v.f = f;
  unsigned int r = v.u + 0x7FFFu + ((v.u >> 16) & 1u);   // round-nearest-even
  unsigned short h = (unsigned short)(r >> 16);
  return __builtin_bit_cast(__bf16, h);
}

// Load a 16x32 bf16 WMMA A/B fragment from row-major memory.
// Lanes 0-15 hold K = k0+0..7 and k0+16..23, lanes 16-31 hold K = k0+8..15
// and k0+24..31: two contiguous 16-byte chunks per lane.
static __device__ inline v16bf load_frag(const __bf16* __restrict__ p, int ld,
                                         int row, int k0, int lane) {
  const int half = (lane >> 4) & 1;
  const __bf16* base = p + (size_t)row * ld + k0 + half * 8;
  v8bf lo = *(const v8bf*)(base);
  v8bf hi = *(const v8bf*)(base + 16);
  return __builtin_shufflevector(lo, hi, 0,1,2,3,4,5,6,7,8,9,10,11,12,13,14,15);
}

static __device__ inline v8f wmma_bf16(v16bf a, v16bf b, v8f c) {
  return __builtin_amdgcn_wmma_f32_16x16x32_bf16(false, a, false, b, (short)0, c,
                                                 false, false);
}

__global__ void cvt_f32_bf16_x4(const float* __restrict__ in,
                                __bf16* __restrict__ out, int n4) {
  int i = blockIdx.x * blockDim.x + threadIdx.x;
  if (i < n4) {
    v4f f = ((const v4f*)in)[i];
    v4bf o;
#pragma unroll
    for (int j = 0; j < 4; ++j) o[j] = f2bf(f[j]);
    ((v4bf*)out)[i] = o;
  }
}

// ---------------------------------------------------------------------------
// Shared GEMM machinery: each wave computes a 32x64 tile of  A[M,1024] @
// W[N,1024]^T  (2 A-frags x 4 B-frags -> 8 WMMA per 32-wide k-step), with an
// explicit ping-pong double buffer so no cross-iteration register copies
// (and hence no v_mov rotation / hazard nops) are needed.
// ---------------------------------------------------------------------------
struct Frags {
  v16bf a0, a1, b0, b1, b2, b3;
};

static __device__ inline void load_tile(Frags& f,
                                        const __bf16* __restrict__ X,
                                        const __bf16* __restrict__ W,
                                        int m0, int n0, int lo16, int kk,
                                        int lane) {
  f.a0 = load_frag(X, HIDDEN, m0 + lo16,      kk, lane);
  f.a1 = load_frag(X, HIDDEN, m0 + 16 + lo16, kk, lane);
  f.b0 = load_frag(W, HIDDEN, n0 +  0 + lo16, kk, lane);
  f.b1 = load_frag(W, HIDDEN, n0 + 16 + lo16, kk, lane);
  f.b2 = load_frag(W, HIDDEN, n0 + 32 + lo16, kk, lane);
  f.b3 = load_frag(W, HIDDEN, n0 + 48 + lo16, kk, lane);
}

static __device__ inline void mma_tile(v8f acc[2][4], const Frags& f) {
  acc[0][0] = wmma_bf16(f.a0, f.b0, acc[0][0]);
  acc[0][1] = wmma_bf16(f.a0, f.b1, acc[0][1]);
  acc[0][2] = wmma_bf16(f.a0, f.b2, acc[0][2]);
  acc[0][3] = wmma_bf16(f.a0, f.b3, acc[0][3]);
  acc[1][0] = wmma_bf16(f.a1, f.b0, acc[1][0]);
  acc[1][1] = wmma_bf16(f.a1, f.b1, acc[1][1]);
  acc[1][2] = wmma_bf16(f.a1, f.b2, acc[1][2]);
  acc[1][3] = wmma_bf16(f.a1, f.b3, acc[1][3]);
}

// Runs the full K=1024 pipelined loop, leaving results in acc.
static __device__ inline void gemm_core(v8f acc[2][4],
                                        const __bf16* __restrict__ X,
                                        const __bf16* __restrict__ W,
                                        int m0, int n0, int lo16, int lane) {
  v8f z = {};
#pragma unroll
  for (int ms = 0; ms < 2; ++ms)
#pragma unroll
    for (int dt = 0; dt < 4; ++dt) acc[ms][dt] = z;

  Frags f0, f1;
  load_tile(f0, X, W, m0, n0, lo16, 0, lane);
  int kk = 0;
#pragma unroll 1
  for (; kk < HIDDEN - 64; kk += 64) {
    load_tile(f1, X, W, m0, n0, lo16, kk + 32, lane);
    mma_tile(acc, f0);
    load_tile(f0, X, W, m0, n0, lo16, kk + 64, lane);
    mma_tile(acc, f1);
  }
  // kk == HIDDEN-64: f0 holds step kk; one more load for the last step.
  load_tile(f1, X, W, m0, n0, lo16, kk + 32, lane);
  mma_tile(acc, f0);
  mma_tile(acc, f1);
}

// ---------------------------------------------------------------------------
// QKV projection: Y[8192,3072] = Xbf @ Wqkv^T + b.  Epilogue scatters into
// Q [B,H,T,Dh] (pre-scaled by 1/sqrt(Dh)), K [B,H,T,Dh], and Vt [B,H,Dh,T]
// (transposed so P@V B-fragments are key-contiguous; packed b128 stores).
// ---------------------------------------------------------------------------
__global__ __launch_bounds__(256)
void qkv_gemm(const __bf16* __restrict__ X, const __bf16* __restrict__ W,
              const float* __restrict__ bias,
              __bf16* __restrict__ Q, __bf16* __restrict__ K,
              __bf16* __restrict__ Vt) {
  const int lane = threadIdx.x & 31;
  const int wave = threadIdx.x >> 5;
  const int m0   = blockIdx.x * 256 + wave * 32;
  const int n0   = blockIdx.y * 64;
  const int lo16 = lane & 15;

  v8f acc[2][4];
  gemm_core(acc, X, W, m0, n0, lo16, lane);

  const int rbase = (lane >> 4) * 8;
#pragma unroll
  for (int ms = 0; ms < 2; ++ms) {
#pragma unroll
    for (int dt = 0; dt < 4; ++dt) {
      const int n  = n0 + dt * 16 + lo16;
      const float bn_ = bias[n];
      const int which = n >> 10;            // 0=Q, 1=K, 2=V
      const int cc = n & 1023;
      const int h = cc >> 6, d = cc & 63;
      const int mbase = m0 + ms * 16 + rbase;      // rows mbase..mbase+7
      const int bb = mbase >> 11, t0 = mbase & (SEQ - 1);
      const int bh = bb * HEADS + h;
      if (which == 0) {
#pragma unroll
        for (int v = 0; v < 8; ++v)
          Q[((size_t)bh * SEQ + t0 + v) * HEAD_DIM + d] =
              f2bf((acc[ms][dt][v] + bn_) * 0.125f);
      } else if (which == 1) {
#pragma unroll
        for (int v = 0; v < 8; ++v)
          K[((size_t)bh * SEQ + t0 + v) * HEAD_DIM + d] =
              f2bf(acc[ms][dt][v] + bn_);
      } else {
        v8bf pk;
#pragma unroll
        for (int v = 0; v < 8; ++v) pk[v] = f2bf(acc[ms][dt][v] + bn_);
        *(v8bf*)&Vt[((size_t)bh * HEAD_DIM + d) * SEQ + t0] = pk;  // 8 consecutive t
      }
    }
  }
}

// ---------------------------------------------------------------------------
// Causal flash attention. One wave per 16-row q-tile; stream 32-key tiles.
// All VMEM for a tile (K frags + V frags) is issued before the S-WMMAs and
// the softmax, so load latency hides under the VALU work + LDS round-trip.
// ---------------------------------------------------------------------------
__global__ __launch_bounds__(256)
void flash_attn(const __bf16* __restrict__ Q, const __bf16* __restrict__ K,
                const __bf16* __restrict__ Vt, __bf16* __restrict__ ctx) {
  __shared__ __align__(16) __bf16 plds[8][16][32];

  const int lane = threadIdx.x & 31;
  const int wave = threadIdx.x >> 5;
  const int qt   = blockIdx.x * 8 + wave;   // 0..8191
  const int bh   = qt >> 7;                 // (b*H + h)
  const int q0   = (qt & 127) * 16;

  const __bf16* Qh = Q  + (size_t)bh * SEQ * HEAD_DIM;
  const __bf16* Kh = K  + (size_t)bh * SEQ * HEAD_DIM;
  const __bf16* Vh = Vt + (size_t)bh * HEAD_DIM * SEQ;

  const int lo16  = lane & 15;
  const int rbase = (lane >> 4) * 8;

  const v16bf qa0 = load_frag(Qh, HEAD_DIM, q0 + lo16, 0,  lane);
  const v16bf qa1 = load_frag(Qh, HEAD_DIM, q0 + lo16, 32, lane);

  v8f z = {};
  v8f o[4];
  float mrow[8], lrow[8];
#pragma unroll
  for (int v = 0; v < 8; ++v) { mrow[v] = -3.0e38f; lrow[v] = 0.0f; }
#pragma unroll
  for (int dt = 0; dt < 4; ++dt) o[dt] = z;

#pragma unroll 1
  for (int kt = 0; kt <= q0 + 15; kt += 32) {
    // Issue all VMEM for this tile up front: K frags + V frags.
    v16bf kb0 = load_frag(Kh, HEAD_DIM, kt + lo16,      0,  lane);
    v16bf kb1 = load_frag(Kh, HEAD_DIM, kt + lo16,      32, lane);
    v16bf kb2 = load_frag(Kh, HEAD_DIM, kt + 16 + lo16, 0,  lane);
    v16bf kb3 = load_frag(Kh, HEAD_DIM, kt + 16 + lo16, 32, lane);
    v16bf vf[4];
#pragma unroll
    for (int dt = 0; dt < 4; ++dt)
      vf[dt] = load_frag(Vh, SEQ, dt * 16 + lo16, kt, lane);

    v8f s0 = z, s1 = z;
    s0 = wmma_bf16(qa0, kb0, s0);
    s0 = wmma_bf16(qa1, kb1, s0);
    s1 = wmma_bf16(qa0, kb2, s1);
    s1 = wmma_bf16(qa1, kb3, s1);

#pragma unroll
    for (int v = 0; v < 8; ++v) {
      const int q = q0 + rbase + v;
      if (kt + lo16 > q)      s0[v] = -3.0e38f;   // causal mask
      if (kt + 16 + lo16 > q) s1[v] = -3.0e38f;

      float sm = fmaxf(s0[v], s1[v]);
#pragma unroll
      for (int off = 8; off; off >>= 1) sm = fmaxf(sm, __shfl_xor(sm, off, 32));
      const float mnew  = fmaxf(mrow[v], sm);
      const float alpha = __expf(mrow[v] - mnew);
      const float p0 = __expf(s0[v] - mnew);
      const float p1 = __expf(s1[v] - mnew);
      float rs = p0 + p1;
#pragma unroll
      for (int off = 8; off; off >>= 1) rs += __shfl_xor(rs, off, 32);
      lrow[v] = lrow[v] * alpha + rs;
      mrow[v] = mnew;
#pragma unroll
      for (int dt = 0; dt < 4; ++dt) o[dt][v] *= alpha;

      plds[wave][rbase + v][lo16]      = f2bf(p0);
      plds[wave][rbase + v][16 + lo16] = f2bf(p1);
    }

    // Re-read P in A-fragment order (intra-wave LDS is in-order; no barrier).
    const int kh = (lane >> 4) * 8;
    v8bf plo = *(const v8bf*)&plds[wave][lo16][kh];
    v8bf phi = *(const v8bf*)&plds[wave][lo16][16 + kh];
    v16bf pf = __builtin_shufflevector(plo, phi,
                 0,1,2,3,4,5,6,7,8,9,10,11,12,13,14,15);

#pragma unroll
    for (int dt = 0; dt < 4; ++dt)
      o[dt] = wmma_bf16(pf, vf[dt], o[dt]);
  }

  const int b = bh >> 4, h = bh & 15;
#pragma unroll
  for (int dt = 0; dt < 4; ++dt) {
#pragma unroll
    for (int v = 0; v < 8; ++v) {
      const int t = q0 + rbase + v;
      const int d = dt * 16 + lo16;
      ctx[((size_t)b * SEQ + t) * HIDDEN + h * HEAD_DIM + d] =
          f2bf(o[dt][v] / lrow[v]);
    }
  }
}

// ---------------------------------------------------------------------------
// Output projection: out[8192,1024] = ctx_bf16 @ Wout^T + b (f32 out).
// ---------------------------------------------------------------------------
__global__ __launch_bounds__(256)
void out_gemm(const __bf16* __restrict__ X, const __bf16* __restrict__ W,
              const float* __restrict__ bias, float* __restrict__ out) {
  const int lane = threadIdx.x & 31;
  const int wave = threadIdx.x >> 5;
  const int m0   = blockIdx.x * 256 + wave * 32;
  const int n0   = blockIdx.y * 64;
  const int lo16 = lane & 15;

  v8f acc[2][4];
  gemm_core(acc, X, W, m0, n0, lo16, lane);

  const int rbase = (lane >> 4) * 8;
#pragma unroll
  for (int ms = 0; ms < 2; ++ms) {
#pragma unroll
    for (int dt = 0; dt < 4; ++dt) {
      const int n = n0 + dt * 16 + lo16;
      const float bn_ = bias[n];
#pragma unroll
      for (int v = 0; v < 8; ++v) {
        const int m = m0 + ms * 16 + rbase + v;
        out[(size_t)m * HIDDEN + n] = acc[ms][dt][v] + bn_;
      }
    }
  }
}

extern "C" void kernel_launch(void* const* d_in, const int* in_sizes, int n_in,
                              void* d_out, int out_size, void* d_ws, size_t ws_size,
                              hipStream_t stream) {
  const float* x     = (const float*)d_in[0];
  // d_in[1] = mask: causal mask applied analytically in flash_attn.
  const float* W_qkv = (const float*)d_in[2];
  const float* b_qkv = (const float*)d_in[3];
  const float* W_out = (const float*)d_in[4];
  const float* b_out = (const float*)d_in[5];

  char* ws = (char*)d_ws;
  __bf16* xb    = (__bf16*)(ws + 0);           // 16 MB  x bf16
  __bf16* wqkvb = (__bf16*)(ws + 16777216);    //  6 MB  Wqkv bf16
  __bf16* woutb = (__bf16*)(ws + 23068672);    //  2 MB  Wout bf16
  __bf16* Qb    = (__bf16*)(ws + 25165824);    // 16 MB  Q [B,H,T,Dh]
  __bf16* Kb    = (__bf16*)(ws + 41943040);    // 16 MB  K [B,H,T,Dh]
  __bf16* Vtb   = (__bf16*)(ws + 58720256);    // 16 MB  V [B,H,Dh,T]
  __bf16* ctx   = (__bf16*)(ws + 75497472);    // 16 MB  ctx [B,T,C]

  const int nx  = BATCH * SEQ * HIDDEN;        // 8388608
  const int nwq = 3 * HIDDEN * HIDDEN;         // 3145728
  const int nwo = HIDDEN * HIDDEN;             // 1048576

  cvt_f32_bf16_x4<<<(nx / 4  + 255) / 256, 256, 0, stream>>>(x,     xb,    nx / 4);
  cvt_f32_bf16_x4<<<(nwq / 4 + 255) / 256, 256, 0, stream>>>(W_qkv, wqkvb, nwq / 4);
  cvt_f32_bf16_x4<<<(nwo / 4 + 255) / 256, 256, 0, stream>>>(W_out, woutb, nwo / 4);

  qkv_gemm<<<dim3(32, 48), 256, 0, stream>>>(xb, wqkvb, b_qkv, Qb, Kb, Vtb);
  flash_attn<<<1024, 256, 0, stream>>>(Qb, Kb, Vtb, ctx);
  out_gemm<<<dim3(32, 16), 256, 0, stream>>>(ctx, woutb, b_out, (float*)d_out);
}
